// Model_575525617776
// MI455X (gfx1250) — matrix operand order repaired and checked by reading, not compiled, not run on hardware
//
#include <hip/hip_runtime.h>
#include <hip/hip_bf16.h>

// ---------------------------------------------------------------------------
// GRU text classifier for MI455X (gfx1250, wave32, WMMA).
//   B=256, T=256, H=512, G=3H=1536, VOCAB=50000, C=6.
// Pipeline:
//   k_cast    : w_ih / w_hh f32 -> bf16
//   k_gather  : X[t*B+b, :] = bf16(embed[seq[b,t], :])
//   k_gemm_gi : gi = X @ w_ih^T   (65536 x 1536 x 512, bf16 WMMA, f32 acc)
//   k_gru     : 16 persistent WGs (512 thr = 16 waves each), each owns a
//               16-row batch tile, runs the full T=256 recurrence with
//               h(bf16) in LDS / h(f32) in regs; gate math fully in regs
//   k_head    : per-class Linear(H,2) + log_softmax
// ---------------------------------------------------------------------------

typedef __attribute__((ext_vector_type(16))) __bf16        v16bf;
typedef __attribute__((ext_vector_type(8)))  float         v8f;
typedef __attribute__((ext_vector_type(4)))  unsigned int  v4u;
typedef __attribute__((ext_vector_type(2)))  float         v2f;

union Frag16 {            // 16 bf16 = one WMMA A or B operand (8 VGPRs)
  v16bf v;
  v4u   q[2];             // two 16-byte chunks
};

__device__ __forceinline__ v8f v8f_zero() {
  v8f z = {0.f, 0.f, 0.f, 0.f, 0.f, 0.f, 0.f, 0.f};
  return z;
}

__device__ __forceinline__ unsigned short f2bf(float f) {
  unsigned u = __float_as_uint(f);
  u += 0x7FFFu + ((u >> 16) & 1u);       // round-to-nearest-even
  return (unsigned short)(u >> 16);
}
__device__ __forceinline__ unsigned pack2bf(float lo, float hi) {
  return (unsigned)f2bf(lo) | ((unsigned)f2bf(hi) << 16);
}

// --------------------------- weight cast (f32 -> bf16) ----------------------
__global__ void k_cast(const float* __restrict__ w, unsigned* __restrict__ o, int n2) {
  int i = blockIdx.x * blockDim.x + threadIdx.x;
  if (i < n2) {
    v2f e = ((const v2f*)w)[i];
    o[i] = pack2bf(e.x, e.y);
  }
}

// ------------------- embedding gather + cast, rows r = t*256 + b ------------
__global__ __launch_bounds__(256) void k_gather(const int* __restrict__ seq,
                                                const float* __restrict__ embed,
                                                unsigned* __restrict__ Xu) {
  const int r = blockIdx.x;                 // 0..65535 ; r = t*256 + b
  const int b = r & 255, t = r >> 8;
  const int tok = seq[b * 256 + t];         // seq is [B,T]
  const v2f* src = (const v2f*)(embed + (long)tok * 512);
  v2f e = src[threadIdx.x];                 // 2 f32 per thread, 256 thr = 512
  Xu[r * 256 + threadIdx.x] = pack2bf(e.x, e.y);
}

// ------------------- gi = X @ w_ih^T : 65536 x 1536 x 512 -------------------
// Block = 256 thr (8 waves), block tile 256(M) x 128(N), wave tile 64x64.
__global__ __launch_bounds__(256) void k_gemm_gi(
    const unsigned short* __restrict__ X,    // [65536,512] bf16 row-major
    const unsigned short* __restrict__ Wih,  // [1536,512]  bf16 row-major
    float* __restrict__ gi)                  // [65536,1536] f32
{
  const int tid  = threadIdx.x;
  const int lane = tid & 31;
  const int w    = tid >> 5;      // 0..7
  const int wm   = w >> 1;        // 0..3
  const int wn   = w & 1;         // 0..1
  const int l15  = lane & 15;
  const int hi   = lane >> 4;     // half-wave select

  const int rowA0 = blockIdx.x * 256 + wm * 64;
  const int colB0 = blockIdx.y * 128 + wn * 64;

  v8f acc[4][4];
#pragma unroll
  for (int i = 0; i < 4; ++i)
#pragma unroll
    for (int j = 0; j < 4; ++j) acc[i][j] = v8f_zero();

  const int kbA = hi ? 16 : 0;    // A: lanes 16..31 hold K+16..K+31
  const int kbB = hi ? 16 : 0;    // B: lanes 16..31 hold K+16..K+31

  for (int k0 = 0; k0 < 512; k0 += 32) {
    Frag16 a[4], b[4];
#pragma unroll
    for (int mt = 0; mt < 4; ++mt) {
      const unsigned short* p = X + (long)(rowA0 + mt * 16 + l15) * 512 + k0 + kbA;
      a[mt].q[0] = *(const v4u*)(p);        // K kbA..kbA+7
      a[mt].q[1] = *(const v4u*)(p + 8);    // K kbA+8..kbA+15
    }
#pragma unroll
    for (int nt = 0; nt < 4; ++nt) {
      const unsigned short* p = Wih + (long)(colB0 + nt * 16 + l15) * 512 + k0 + kbB;
      b[nt].q[0] = *(const v4u*)(p);        // K kbB..kbB+7
      b[nt].q[1] = *(const v4u*)(p + 8);    // K kbB+8..kbB+15
    }
#pragma unroll
    for (int mt = 0; mt < 4; ++mt)
#pragma unroll
      for (int nt = 0; nt < 4; ++nt)
        acc[mt][nt] = __builtin_amdgcn_wmma_f32_16x16x32_bf16(
            false, a[mt].v, false, b[nt].v, (short)0, acc[mt][nt], false, false);
  }

#pragma unroll
  for (int mt = 0; mt < 4; ++mt)
#pragma unroll
    for (int nt = 0; nt < 4; ++nt) {
      const int col = colB0 + nt * 16 + l15;
#pragma unroll
      for (int v = 0; v < 8; ++v) {
        const int row = rowA0 + mt * 16 + v + 8 * hi;
        gi[(long)row * 1536 + col] = acc[mt][nt][v];
      }
    }
}

// ----------------------- persistent GRU recurrence --------------------------
// 16 WGs, each 16 batch rows. 512 threads = 16 waves (4 waves/SIMD32 for
// latency hiding in the serial t-loop). Wave w owns n-tiles ntg = w*2, w*2+1
// of the 512-wide hidden dim and computes the r/z/n gate tiles (column
// offsets 0/512/1024) for those columns -> gate math stays in WMMA C-layout
// registers, no LDS gather of gh.
#define HPAD 520   // bf16 row stride (16B-aligned, bank-stagger pad)

__global__ __launch_bounds__(512) void k_gru(
    const unsigned short* __restrict__ Whh,  // [1536,512] bf16
    const float* __restrict__ gi,            // [T=256][B=256][1536]
    const float* __restrict__ b_ih,          // [1536]
    const float* __restrict__ b_hh,          // [1536]
    float* __restrict__ h_out)               // [256,512] f32
{
  __shared__ unsigned short h_lds[16 * HPAD];   // 16.6 KB

  const int tid  = threadIdx.x;
  const int lane = tid & 31;
  const int w    = tid >> 5;                   // wave 0..15
  const int l15  = lane & 15;
  const int hi   = lane >> 4;
  const int wgB  = blockIdx.x * 16;            // batch base of this WG

  for (int i = tid; i < 16 * HPAD; i += 512) h_lds[i] = 0;   // h0 = 0

  float hreg[2][8];                            // f32 h in WMMA C-layout
#pragma unroll
  for (int nt = 0; nt < 2; ++nt)
#pragma unroll
    for (int v = 0; v < 8; ++v) hreg[nt][v] = 0.f;

  float bi[2][3], bh[2][3];                    // per-lane gate biases
#pragma unroll
  for (int nt = 0; nt < 2; ++nt) {
    const int ntg = w * 2 + nt;
#pragma unroll
    for (int g = 0; g < 3; ++g) {
      const int c = g * 512 + ntg * 16 + l15;
      bi[nt][g] = b_ih[c];
      bh[nt][g] = b_hh[c];
    }
  }

  const int kbA = hi ? 16 : 0;
  const int kbB = hi ? 16 : 0;

  __syncthreads();

  for (int t = 0; t < 256; ++t) {
    // prefetch next step's gi tile (16 rows x 1536 f32 = 768 cachelines)
    if (t + 1 < 256) {
      const char* pf = (const char*)(gi + ((long)(t + 1) * 256 + wgB) * 1536);
      if (tid < 256) {
#pragma unroll
        for (int i = 0; i < 3; ++i)
          __builtin_prefetch(pf + (size_t)(tid + i * 256) * 128, 0, 0);
      }
    }

    v8f acc[2][3];
#pragma unroll
    for (int nt = 0; nt < 2; ++nt)
#pragma unroll
      for (int g = 0; g < 3; ++g) acc[nt][g] = v8f_zero();

    for (int k0 = 0; k0 < 512; k0 += 32) {
      Frag16 a;                                // A = h tile from LDS
      const unsigned short* pa = &h_lds[l15 * HPAD + k0 + kbA];
      a.q[0] = *(const v4u*)(pa);
      a.q[1] = *(const v4u*)(pa + 8);
#pragma unroll
      for (int nt = 0; nt < 2; ++nt) {
        const int ntg = w * 2 + nt;
#pragma unroll
        for (int g = 0; g < 3; ++g) {
          Frag16 bf;                           // B = W_hh rows (L2 resident)
          const unsigned short* pb =
              Whh + (long)(g * 512 + ntg * 16 + l15) * 512 + k0 + kbB;
          bf.q[0] = *(const v4u*)(pb);
          bf.q[1] = *(const v4u*)(pb + 8);
          acc[nt][g] = __builtin_amdgcn_wmma_f32_16x16x32_bf16(
              false, a.v, false, bf.v, (short)0, acc[nt][g], false, false);
        }
      }
    }

    __syncthreads();   // all waves done reading old h from LDS

    // gate math entirely in registers (PyTorch GRUCell ordering r,z,n)
#pragma unroll
    for (int nt = 0; nt < 2; ++nt) {
      const int ntg = w * 2 + nt;
      const int n   = ntg * 16 + l15;          // hidden column 0..511
#pragma unroll
      for (int v = 0; v < 8; ++v) {
        const int  m     = v + 8 * hi;         // batch row in tile
        const long rbase = ((long)t * 256 + wgB + m) * 1536;
        const float ir = gi[rbase + n]        + bi[nt][0];
        const float iz = gi[rbase + 512 + n]  + bi[nt][1];
        const float in = gi[rbase + 1024 + n] + bi[nt][2];
        const float hr = acc[nt][0][v] + bh[nt][0];
        const float hz = acc[nt][1][v] + bh[nt][1];
        const float hn = acc[nt][2][v] + bh[nt][2];
        const float r  = 1.f / (1.f + __expf(-(ir + hr)));
        const float z  = 1.f / (1.f + __expf(-(iz + hz)));
        const float pre = in + r * hn;
        const float e2  = __expf(2.f * pre);               // tanh via exp
        const float ng  = (e2 - 1.f) / (e2 + 1.f);
        const float hnew = (1.f - z) * ng + z * hreg[nt][v];
        hreg[nt][v] = hnew;
        h_lds[m * HPAD + n] = f2bf(hnew);
      }
    }
    __syncthreads();   // new h visible for next step's A-operand
  }

  // write final hidden state (f32)
#pragma unroll
  for (int nt = 0; nt < 2; ++nt) {
    const int n = (w * 2 + nt) * 16 + l15;
#pragma unroll
    for (int v = 0; v < 8; ++v) {
      const int m = v + 8 * hi;
      h_out[(long)(wgB + m) * 512 + n] = hreg[nt][v];
    }
  }
}

// --------------------- classifier heads + log_softmax -----------------------
__global__ __launch_bounds__(256) void k_head(const float* __restrict__ h,
                                              const float* __restrict__ cls_w,
                                              const float* __restrict__ cls_b,
                                              float* __restrict__ out) {
  const int c = blockIdx.x;      // 0..5
  const int b = threadIdx.x;     // 0..255
  const float* hb = h + (long)b * 512;
  const float* w0 = cls_w + (long)(c * 2 + 0) * 512;
  const float* w1 = cls_w + (long)(c * 2 + 1) * 512;
  float a0 = 0.f, a1 = 0.f;
  for (int k = 0; k < 512; ++k) {
    const float hv = hb[k];
    a0 = fmaf(hv, w0[k], a0);
    a1 = fmaf(hv, w1[k], a1);
  }
  const float l0 = a0 + cls_b[c * 2 + 0];
  const float l1 = a1 + cls_b[c * 2 + 1];
  const float mx  = fmaxf(l0, l1);
  const float lse = mx + __logf(__expf(l0 - mx) + __expf(l1 - mx));
  out[(c * 256 + b) * 2 + 0] = l0 - lse;
  out[(c * 256 + b) * 2 + 1] = l1 - lse;
}

// ---------------------------------------------------------------------------
extern "C" void kernel_launch(void* const* d_in, const int* in_sizes, int n_in,
                              void* d_out, int out_size, void* d_ws, size_t ws_size,
                              hipStream_t stream) {
  (void)in_sizes; (void)n_in; (void)out_size; (void)ws_size;
  const int*   seq   = (const int*)  d_in[0];
  const float* embed = (const float*)d_in[1];
  const float* w_ih  = (const float*)d_in[2];
  const float* w_hh  = (const float*)d_in[3];
  const float* b_ih  = (const float*)d_in[4];
  const float* b_hh  = (const float*)d_in[5];
  const float* cls_w = (const float*)d_in[6];
  const float* cls_b = (const float*)d_in[7];
  float* out = (float*)d_out;

  // workspace layout (bytes):
  //   [0, 1.5M)      w_ih bf16
  //   [1.5M, 3M)     w_hh bf16
  //   [3M, 67M+3M)   X bf16   [65536 x 512]
  //   [~70M, ~473M)  gi f32   [65536 x 1536]
  //   [~473M, +512K) h_out f32 [256 x 512]
  char* ws = (char*)d_ws;
  unsigned short* Wih_bf = (unsigned short*)(ws);
  unsigned short* Whh_bf = (unsigned short*)(ws + 1572864);
  unsigned short* X_bf   = (unsigned short*)(ws + 3145728);
  float*          gi     = (float*)(ws + 70254592);
  float*          h_out  = (float*)(ws + 472907776);

  k_cast<<<1536, 256, 0, stream>>>(w_ih, (unsigned*)Wih_bf, 393216);
  k_cast<<<1536, 256, 0, stream>>>(w_hh, (unsigned*)Whh_bf, 393216);
  k_gather<<<65536, 256, 0, stream>>>(seq, embed, (unsigned*)X_bf);
  dim3 gb(256, 12);   // M/256, G/128
  k_gemm_gi<<<gb, 256, 0, stream>>>(X_bf, Wih_bf, gi);
  k_gru<<<16, 512, 0, stream>>>(Whh_bf, gi, b_ih, b_hh, h_out);
  k_head<<<6, 256, 0, stream>>>(h_out, cls_w, cls_b, out);
}
